// Ds2Gru_4372276707435
// MI455X (gfx1250) — compile-verified, matrix operand order checked by reading, not run on hardware
//
#include <hip/hip_runtime.h>
#include <hip/hip_bf16.h>

// ---------------------------------------------------------------------------
// BiGRU (3 layers) + BatchNorm for MI455X / gfx1250 (wave32, WMMA, TDM).
//   1) Convert x and all weights to bf16 (deterministic, every call).
//   2) Input projections xg = x @ W_ih^T + b_ih as bf16 WMMA GEMM,
//      M=32 blocking so each B fragment feeds two WMMAs (halves L2 traffic).
//   3) Recurrence: weight-stationary. Each chain (dir, 16-batch tile) is
//      split across 16 workgroups, each owning a gate-aligned 96-column
//      W_hh slice resident in LDS (~100KB of 320KB WGP LDS), loaded via
//      the Tensor Data Mover with hardware LDS padding (520-bf16 rows).
//      Per step: TDM reload of the 16KB h tile, WMMA gh-slice, fp32 gate
//      math, publish h via L2 with per-step release/acquire counters.
//   4) BatchNorm: atomic channel stats + elementwise apply (fp32).
// ---------------------------------------------------------------------------

typedef __bf16 bf16;
typedef __attribute__((ext_vector_type(16))) __bf16 v16bf;
typedef __attribute__((ext_vector_type(8)))  float  v8f;
typedef __attribute__((ext_vector_type(4)))  unsigned int u32x4;
typedef __attribute__((ext_vector_type(8)))  int i32x8;
typedef __attribute__((ext_vector_type(4)))  int i32x4;

static constexpr int kB  = 32;
static constexpr int kT  = 512;
static constexpr int kBT = kB * kT;      // 16384
static constexpr int kH  = 512;
static constexpr int kH3 = 3 * kH;       // 1536
static constexpr int kH2 = 2 * kH;       // 1024

// Recurrence sharding
static constexpr int NSPLIT = 16;        // workgroups per chain (N-shards)
static constexpr int JW     = kH / NSPLIT;   // 32 hidden units per shard
static constexpr int NCOL   = 3 * JW;        // 96 gate columns per shard
static constexpr int WLD    = 520;       // padded LDS row (bf16): 512 + 16B pad
static constexpr int HLD    = 520;
static constexpr int GHLD   = 100;

union Frag16 { v16bf v; uint4 q[2]; };

// Load a 16x32 bf16 WMMA A/B fragment from a row-major [rows][ld] matrix.
// Per ISA layout: lanes 0-15 hold row (lane&15), chunks K=[k0..k0+7] and
// [k0+16..k0+23]; lanes 16-31 hold chunks shifted by +8. Two 16B loads/lane.
__device__ inline v16bf load_frag(const bf16* base, int ld, int row0, int k0,
                                  int lane) {
    const int r  = row0 + (lane & 15);
    const int kc = k0 + ((lane >> 4) << 3);
    const bf16* p = base + (size_t)r * ld + kc;
    Frag16 f;
    f.q[0] = *(const uint4*)(p);
    f.q[1] = *(const uint4*)(p + 16);
    return f.v;
}

__device__ inline v8f wmma_bf16(v16bf a, v16bf b, v8f c) {
    return __builtin_amdgcn_wmma_f32_16x16x32_bf16(
        /*neg_a=*/false, a, /*neg_b=*/false, b,
        /*c_mod=*/(short)0, c, /*reuse_a=*/false, /*reuse_b=*/false);
}

__device__ inline float sigmoidf_(float x) {
    return 1.0f / (1.0f + __expf(-x));
}

// ---------------------------------------------------------------------------
// Tensor Data Mover: 2D bf16 tile load global->LDS with optional LDS padding.
// D# per CDNA5 ISA:
//  group0: [1:0]=count=1, [63:32]=lds_addr, [120:64]=global_addr, [127:126]=2
//  group1: [17:16]=data_size(1=2B), [20]=pad_en, [24:22]=pad_interval,
//          [31:25]=pad_amount, [79:48]=tensor_dim0, [111:80]=tensor_dim1,
//          [127:112]=tile_dim0, [143:128]=tile_dim1, [207:160]=dim0_stride
//  groups 2/3 + trailing group: zero (2D tile).
// pad_interval=7 -> every 256 DWORDs (1024B row), pad_amount=3 -> 4 DWORDs,
// i.e. 512-element bf16 rows land on 520-element padded LDS rows.
// This toolchain exposes the 6-arg builtin (clang-23 / therock-10.0 form).
// ---------------------------------------------------------------------------
__device__ inline void tdm_load_2d_bf16(unsigned lds_addr, const void* gptr,
                                        unsigned tile_cols, unsigned tile_rows,
                                        unsigned tensor_d0, unsigned tensor_d1,
                                        unsigned long long d0_stride,
                                        unsigned pad_en, unsigned pad_interval,
                                        unsigned pad_amount) {
    const unsigned long long ga = (unsigned long long)(size_t)gptr;
    u32x4 g0;
    g0[0] = 1u;                                            // count=1
    g0[1] = lds_addr;
    g0[2] = (unsigned)(ga & 0xffffffffu);
    g0[3] = (unsigned)((ga >> 32) & 0x1ffffffu) | (2u << 30);  // type=2
    i32x8 g1;
    g1[0] = (int)((1u << 16) | (pad_en << 20) | (pad_interval << 22) |
                  (pad_amount << 25));
    g1[1] = (int)((tensor_d0 & 0xffffu) << 16);
    g1[2] = (int)(((tensor_d0 >> 16) & 0xffffu) |
                  ((tensor_d1 & 0xffffu) << 16));
    g1[3] = (int)(((tensor_d1 >> 16) & 0xffffu) |
                  ((tile_cols & 0xffffu) << 16));
    g1[4] = (int)(tile_rows & 0xffffu);                    // tile_dim1 (dim2=0)
    g1[5] = (int)(unsigned)(d0_stride & 0xffffffffull);
    g1[6] = (int)(unsigned)((d0_stride >> 32) & 0xffffull);
    g1[7] = 0;
    const i32x4 z4 = {0, 0, 0, 0};
    const i32x8 z8 = {0, 0, 0, 0, 0, 0, 0, 0};
    __builtin_amdgcn_tensor_load_to_lds(g0, g1, z4, z4, z8, 0);
}

// ---------------------------------------------------------------------------
// fp32 -> bf16 conversion (grid-stride)
// ---------------------------------------------------------------------------
__global__ void k_cvt_bf16(const float* __restrict__ src, bf16* __restrict__ dst,
                           int n) {
    for (int i = blockIdx.x * blockDim.x + threadIdx.x; i < n;
         i += gridDim.x * blockDim.x)
        dst[i] = (bf16)src[i];
}

// ---------------------------------------------------------------------------
// Input projection GEMM: xg[dir][m][n] = b_ih[dir][n] + sum_k x[m][k]*W[dir][n][k]
// grid = (BT/32, 2), block = 256 (8 waves). Wave w owns cols [w*192, +192),
// two 16-row M-halves share each B fragment (B traffic halved).
// ---------------------------------------------------------------------------
__global__ void __launch_bounds__(256) k_gemm_xg(
    const bf16* __restrict__ x,      // [BT][K]
    const bf16* __restrict__ wih,    // [2][1536][K]
    const float* __restrict__ bih,   // [2][1536]
    float* __restrict__ xg,          // [2][BT][1536]
    int K) {
    const int lane = threadIdx.x & 31;
    const int wave = threadIdx.x >> 5;
    const int m0   = blockIdx.x * 32;
    const int dir  = blockIdx.y;
    const bf16*  W   = wih + (size_t)dir * kH3 * K;
    const float* bb  = bih + dir * kH3;
    float*       out = xg + ((size_t)dir * kBT + m0) * kH3;
    const int n0 = wave * 192;

    v8f acc[2][12] = {};
    for (int k0 = 0; k0 < K; k0 += 32) {
        v16bf a0 = load_frag(x, K, m0, k0, lane);
        v16bf a1 = load_frag(x, K, m0 + 16, k0, lane);
        #pragma unroll
        for (int nt = 0; nt < 12; ++nt) {
            v16bf b = load_frag(W, K, n0 + nt * 16, k0, lane);
            acc[0][nt] = wmma_bf16(a0, b, acc[0][nt]);
            acc[1][nt] = wmma_bf16(a1, b, acc[1][nt]);
        }
    }
    const int ml = (lane >> 4) * 8;
    const int nl = lane & 15;
    #pragma unroll
    for (int mh = 0; mh < 2; ++mh) {
        #pragma unroll
        for (int nt = 0; nt < 12; ++nt) {
            const int n = n0 + nt * 16 + nl;
            const float bv = bb[n];
            #pragma unroll
            for (int r = 0; r < 8; ++r)
                out[(size_t)(mh * 16 + ml + r) * kH3 + n] = acc[mh][nt][r] + bv;
        }
    }
}

// ---------------------------------------------------------------------------
// GRU recurrence, weight-stationary LDS shards, TDM-fed.
// grid = (NSPLIT, 4 chains), block = 64 (2 waves).
// ---------------------------------------------------------------------------
__global__ void __launch_bounds__(64) k_gru_seq(
    const float* __restrict__ xg,   // [2][BT][1536]
    const bf16*  __restrict__ whh,  // [2][1536][512]
    const float* __restrict__ bhh,  // [2][1536]
    float* __restrict__ y,          // [B][T][1024]
    bf16* __restrict__ hbuf,        // [2 parity][4 chain][16][512]
    int* __restrict__ cnt) {        // [4][512]
    __shared__ bf16  wlds[NCOL][WLD];   // weight slice (N-major, TDM-padded)
    __shared__ bf16  hlds[16][HLD];     // full h (bf16) for WMMA A
    __shared__ float ghlds[16][GHLD];   // gh slice
    __shared__ float hflds[16][JW];     // fp32 h slice (owned units)
    __shared__ float bhl[NCOL];

    const int tid   = threadIdx.x;
    const int lane  = tid & 31;
    const int wave  = tid >> 5;          // 0..1
    const int split = blockIdx.x;        // 0..NSPLIT-1
    const int chain = blockIdx.y;        // 0..3
    const int dir   = chain >> 1;
    const int mtile = chain & 1;
    const int j0    = split * JW;
    const int b0    = mtile * 16;

    const bf16* W = whh + (size_t)dir * kH3 * kH;
    const unsigned wlds_base = (unsigned)(size_t)&wlds[0][0];
    const unsigned hlds_base = (unsigned)(size_t)&hlds[0][0];

    // TDM preload of the gate-aligned weight slice: 3 tiles of 32 rows x 512,
    // padded by the TDM into 520-bf16 LDS rows. Wave-uniform guard (TDM
    // ignores EXEC, so wave 1 must branch around the issue entirely).
    if (wave == 0) {
        #pragma unroll
        for (int g = 0; g < 3; ++g) {
            tdm_load_2d_bf16(wlds_base + (unsigned)(g * JW) * (WLD * 2),
                             W + (size_t)(g * kH + j0) * kH,
                             /*tile_cols=*/kH, /*tile_rows=*/JW,
                             /*tensor_d0=*/kH, /*tensor_d1=*/JW,
                             /*d0_stride=*/kH,
                             /*pad_en=*/1, /*pad_interval=*/7, /*pad_amount=*/3);
        }
        __builtin_amdgcn_s_wait_tensorcnt(0);
    }
    for (int c = tid; c < NCOL; c += 64) {
        const int g = c / JW, j = c % JW;
        bhl[c] = bhh[dir * kH3 + g * kH + j0 + j];
    }
    for (int i = tid; i < 16 * HLD; i += 64) ((bf16*)hlds)[i] = (bf16)0.0f;
    __syncthreads();

    int*  mycnt = cnt + chain * kT;
    bf16* hb0 = hbuf + (size_t)chain * 16 * kH;            // parity 0
    bf16* hb1 = hbuf + (size_t)(4 + chain) * 16 * kH;      // parity 1

    for (int ti = 0; ti < kT; ++ti) {
        const int t = dir ? (kT - 1 - ti) : ti;
        if (ti > 0) {
            // Acquire: wait for all shards of this chain to publish step ti-1.
            if (tid == 0) {
                while (__hip_atomic_load(&mycnt[ti - 1], __ATOMIC_ACQUIRE,
                                         __HIP_MEMORY_SCOPE_AGENT) < NSPLIT)
                    __builtin_amdgcn_s_sleep(1);
            }
            __syncthreads();
            __threadfence();
            const bf16* hs = ((ti - 1) & 1) ? hb1 : hb0;
            // TDM reload of the full 16x512 h tile into padded LDS rows.
            if (wave == 0) {
                tdm_load_2d_bf16(hlds_base, hs,
                                 /*tile_cols=*/kH, /*tile_rows=*/16,
                                 /*tensor_d0=*/kH, /*tensor_d1=*/16,
                                 /*d0_stride=*/kH, 1, 7, 3);
                __builtin_amdgcn_s_wait_tensorcnt(0);
            }
            __syncthreads();
        }

        // gh[m][c] = sum_k h[m][k] * Wslice[c][k]  (WMMA, fp32 accum)
        v8f acc[3] = {};
        #pragma unroll
        for (int kt = 0; kt < 16; ++kt) {
            v16bf a = load_frag(&hlds[0][0], HLD, 0, kt * 32, lane);
            #pragma unroll
            for (int nt = 0; nt < 3; ++nt) {
                v16bf b = load_frag(&wlds[0][0], WLD, wave * 48 + nt * 16,
                                    kt * 32, lane);
                acc[nt] = wmma_bf16(a, b, acc[nt]);
            }
        }
        const int ml = (lane >> 4) * 8;
        const int nl = lane & 15;
        #pragma unroll
        for (int nt = 0; nt < 3; ++nt) {
            const int c = wave * 48 + nt * 16 + nl;
            #pragma unroll
            for (int r = 0; r < 8; ++r) ghlds[ml + r][c] = acc[nt][r];
        }
        __syncthreads();

        // Gate math + h update for owned units (16 rows x JW units).
        bf16* hw = (ti & 1) ? hb1 : hb0;
        #pragma unroll
        for (int u = 0; u < 8; ++u) {
            const int unit = u * 64 + tid;
            const int m = unit >> 5;        // 0..15
            const int j = unit & (JW - 1);  // 0..31
            const float* xgp =
                xg + ((size_t)dir * kBT + (size_t)(b0 + m) * kT + t) * kH3;
            const float xr = xgp[j0 + j];
            const float xz = xgp[kH + j0 + j];
            const float xn = xgp[2 * kH + j0 + j];
            const float hr = ghlds[m][j]          + bhl[j];
            const float hz = ghlds[m][JW + j]     + bhl[JW + j];
            const float hn = ghlds[m][2 * JW + j] + bhl[2 * JW + j];
            const float r = sigmoidf_(xr + hr);
            const float z = sigmoidf_(xz + hz);
            const float n = tanhf(xn + r * hn);
            const float hprev = (ti == 0) ? 0.0f : hflds[m][j];
            const float hnew = (1.0f - z) * n + z * hprev;
            hflds[m][j] = hnew;
            y[((size_t)(b0 + m) * kT + t) * kH2 + dir * kH + j0 + j] = hnew;
            hw[m * kH + j0 + j] = (bf16)hnew;
        }
        __threadfence();
        __syncthreads();
        if (tid == 0)
            __hip_atomic_fetch_add(&mycnt[ti], 1, __ATOMIC_RELEASE,
                                   __HIP_MEMORY_SCOPE_AGENT);
    }
}

// ---------------------------------------------------------------------------
// BatchNorm: per-channel sum / sumsq via atomics, then elementwise apply.
// ---------------------------------------------------------------------------
__global__ void __launch_bounds__(256) k_bn_stats(const float* __restrict__ y,
                                                  float* __restrict__ stats) {
    const int c0 = threadIdx.x;   // 4 channels per thread: c0, c0+256, ...
    float s[4] = {}, ss[4] = {};
    for (int row = blockIdx.x; row < kBT; row += gridDim.x) {
        const float* p = y + (size_t)row * kH2;
        #pragma unroll
        for (int i = 0; i < 4; ++i) {
            const float v = p[c0 + i * 256];
            s[i] += v; ss[i] += v * v;
        }
    }
    #pragma unroll
    for (int i = 0; i < 4; ++i) {
        atomicAdd(&stats[c0 + i * 256], s[i]);
        atomicAdd(&stats[kH2 + c0 + i * 256], ss[i]);
    }
}

__global__ void __launch_bounds__(256) k_bn_apply(
    float* __restrict__ y, const float* __restrict__ stats,
    const float* __restrict__ gamma, const float* __restrict__ beta) {
    const float invN = 1.0f / (float)kBT;
    for (size_t i = (size_t)blockIdx.x * blockDim.x + threadIdx.x;
         i < (size_t)kBT * kH2; i += (size_t)gridDim.x * blockDim.x) {
        const int c = (int)(i & (kH2 - 1));
        const float mean = stats[c] * invN;
        const float var  = stats[kH2 + c] * invN - mean * mean;
        const float rstd = rsqrtf(var + 1e-5f);
        y[i] = (y[i] - mean) * rstd * gamma[c] + beta[c];
    }
}

// ---------------------------------------------------------------------------
// Host side
// ---------------------------------------------------------------------------
extern "C" void kernel_launch(void* const* d_in, const int* in_sizes, int n_in,
                              void* d_out, int out_size, void* d_ws,
                              size_t ws_size, hipStream_t stream) {
    const float* x     = (const float*)d_in[0];
    const float* w_ih0 = (const float*)d_in[1];
    const float* w_hh0 = (const float*)d_in[2];
    const float* b_ih0 = (const float*)d_in[3];
    const float* b_hh0 = (const float*)d_in[4];
    const float* w_ihr = (const float*)d_in[5];
    const float* w_hhr = (const float*)d_in[6];
    const float* b_ihr = (const float*)d_in[7];
    const float* b_hhr = (const float*)d_in[8];
    const float* gamma = (const float*)d_in[9];
    const float* beta  = (const float*)d_in[10];
    (void)in_sizes; (void)n_in; (void)out_size; (void)ws_size;

    char* ws = (char*)d_ws;
    bf16* WIH0 = (bf16*)ws;  ws += (size_t)2 * kH3 * 1280 * sizeof(bf16);
    bf16* WIHR = (bf16*)ws;  ws += (size_t)2 * 2 * kH3 * kH2 * sizeof(bf16);
    bf16* WHH  = (bf16*)ws;  ws += (size_t)3 * 2 * kH3 * kH * sizeof(bf16);
    bf16* XBF  = (bf16*)ws;  ws += (size_t)kBT * 1280 * sizeof(bf16);
    bf16* HBUF = (bf16*)ws;  ws += (size_t)2 * 4 * 16 * kH * sizeof(bf16);
    float* XG  = (float*)ws; ws += (size_t)2 * kBT * kH3 * sizeof(float);
    float* Y   = (float*)ws; ws += (size_t)kBT * kH2 * sizeof(float);
    float* STATS = (float*)ws; ws += (size_t)2 * kH2 * sizeof(float);
    int*   CNT   = (int*)ws;   ws += (size_t)4 * kT * sizeof(int);

    auto cvt = [&](const float* s, bf16* d, int n) {
        int g = (n + 255) / 256; if (g > 4096) g = 4096;
        k_cvt_bf16<<<g, 256, 0, stream>>>(s, d, n);
    };

    cvt(w_ih0, WIH0, 2 * kH3 * 1280);
    cvt(w_ihr, WIHR, 2 * 2 * kH3 * kH2);
    cvt(w_hh0, WHH, 2 * kH3 * kH);
    cvt(w_hhr, WHH + (size_t)2 * kH3 * kH, 2 * 2 * kH3 * kH);
    cvt(x, XBF, kBT * 1280);

    for (int layer = 0; layer < 3; ++layer) {
        const int K = layer ? kH2 : 1280;
        const bf16* wih = layer ? (WIHR + (size_t)(layer - 1) * 2 * kH3 * kH2)
                                : WIH0;
        const bf16* whh = WHH + (size_t)layer * 2 * kH3 * kH;
        const float* bih = layer ? (b_ihr + (size_t)(layer - 1) * 2 * kH3)
                                 : b_ih0;
        const float* bhh = layer ? (b_hhr + (size_t)(layer - 1) * 2 * kH3)
                                 : b_hh0;
        float* yout = (layer == 2) ? (float*)d_out : Y;

        k_gemm_xg<<<dim3(kBT / 32, 2), 256, 0, stream>>>(XBF, wih, bih, XG, K);

        (void)hipMemsetAsync(CNT, 0, (size_t)4 * kT * sizeof(int), stream);
        k_gru_seq<<<dim3(NSPLIT, 4), 64, 0, stream>>>(XG, whh, bhh, yout,
                                                      HBUF, CNT);

        (void)hipMemsetAsync(STATS, 0, (size_t)2 * kH2 * sizeof(float), stream);
        k_bn_stats<<<256, 256, 0, stream>>>(yout, STATS);
        k_bn_apply<<<2048, 256, 0, stream>>>(yout, STATS,
                                             gamma + (size_t)layer * kH2,
                                             beta + (size_t)layer * kH2);
        if (layer < 2) cvt(yout, XBF, kBT * kH2);
    }
}
